// KGAtt_58153857187903
// MI455X (gfx1250) — matrix operand order repaired and checked by reading, not compiled, not run on hardware
//
#include <hip/hip_runtime.h>
#include <hip/hip_bf16.h>

// ---------------------------------------------------------------------------
// KG attention: gather->concat -> GEMM(768x256, bf16 WMMA) -> score ->
// exp -> global sum -> segment-sum scatter -> elu normalize.
// Memory-bound (~1.7GB traffic vs ~200 GFLOP) -> gather straight into WMMA
// A-fragment layout (prefetched one k-tile ahead); W tiles DMA'd into LDS via
// the Tensor Data Mover, double-buffered, shared by 4 waves per block; B
// fragments ping-pong in registers so ds latency overlaps the WMMA chain.
// ---------------------------------------------------------------------------

typedef __attribute__((ext_vector_type(16))) __bf16        v16bf;
typedef __attribute__((ext_vector_type(8)))  float         v8f;
typedef __attribute__((ext_vector_type(8)))  unsigned int  v8u;
typedef __attribute__((ext_vector_type(4)))  unsigned int  v4u;
typedef __attribute__((ext_vector_type(8)))  int           v8i;
typedef __attribute__((ext_vector_type(4)))  int           v4i;

#define N_ENT    100000
#define IN_DIM   256
#define OUT_DIM  256
#define N_EDGES  500000
#define K_DIM    768          // 3 * IN_DIM
#define K_TILES  24           // 768 / 32
#define N_TILES  16           // 256 / 16
#define TILE_DWORDS (N_TILES * 32 * 8)           // 4096 dwords = 16KB per k-tile
#define WB_DWORDS (K_TILES * TILE_DWORDS)        // 98304 dwords = 393KB

__device__ __forceinline__ unsigned int pack2_bf16(float a, float b) {
  union { __hip_bfloat162 h; unsigned int u; } cv;
  cv.h = __float22bfloat162_rn(make_float2(a, b));   // v_cvt_pk_bf16_f32
  return cv.u;
}

__device__ __forceinline__ void load_bfrag(const unsigned int* p, v8u& dst) {
  const uint4* q = (const uint4*)p;
  uint4 lo = q[0], hi = q[1];                        // 2x ds_load_b128
  dst[0] = lo.x; dst[1] = lo.y; dst[2] = lo.z; dst[3] = lo.w;
  dst[4] = hi.x; dst[5] = hi.y; dst[6] = hi.z; dst[7] = hi.w;
}

// TDM: contiguous 16KB (2048 x 8B) global -> LDS, descriptor per ISA ch.8.
// 6-arg builtin on this toolchain:
//   (uint32x4 g0, int32x8 g1, int32x4 g2, int32x4 g3, int32x8 extra, i32 cpol)
__device__ __forceinline__ void tdm_load_16k(const unsigned int* gptr, unsigned lds_off) {
  unsigned long long ga = (unsigned long long)(size_t)gptr;
  v4u g0;
  g0[0] = 1u;                                 // count=1, user descriptor
  g0[1] = lds_off;                            // lds_addr (bytes)
  g0[2] = (unsigned int)ga;                   // global_addr[31:0]
  g0[3] = (unsigned int)(ga >> 32) | (2u << 30);  // global_addr[56:32] | type=2
  v8i g1;
  g1[0] = (int)(3u << 16);                    // workgroup_mask=0, data_size=3 (8B)
  g1[1] = (int)(2048u << 16);                 // tensor_dim0[15:0] = 2048
  g1[2] = (int)(1u << 16);                    // tensor_dim0 hi=0 | tensor_dim1 = 1
  g1[3] = (int)(2048u << 16);                 // tensor_dim1 hi=0 | tile_dim0 = 2048
  g1[4] = 1;                                  // tile_dim1 = 1, tile_dim2 = 0
  g1[5] = 2048;                               // tensor_dim0_stride = 2048
  g1[6] = 0;
  g1[7] = 0;
  v4i z4 = {0, 0, 0, 0};                      // groups 2/3 unused (2D tensor)
  v8i z8 = {0, 0, 0, 0, 0, 0, 0, 0};
  __builtin_amdgcn_tensor_load_to_lds(g0, g1, z4, z4, z8, 0);
}

// --------------------------- kernel 1: zero accumulators --------------------
__global__ void k_zero(float4* __restrict__ out4, float* __restrict__ e_total, int n4) {
  int i = blockIdx.x * blockDim.x + threadIdx.x;
  if (i < n4) out4[i] = make_float4(0.f, 0.f, 0.f, 0.f);
  if (i == 0) *e_total = 0.f;
}

// --------------------------- kernel 2: pack W_a into B-fragment layout ------
// Wb layout: [ktile(24)][ntile(16)][lane(32)][pair(8)] dwords; each dword is
// two bf16 for K = ktile*32 + (pair>>2)*16 + (lane>=16 ? 8 : 0) + (pair&3)*2, N = ntile*16+lane%16.
__global__ void k_pack(const float* __restrict__ W_a, unsigned int* __restrict__ Wb) {
  int i = blockIdx.x * blockDim.x + threadIdx.x;
  if (i >= WB_DWORDS) return;
  int pair  = i & 7;
  int lane  = (i >> 3) & 31;
  int ntile = (i >> 8) & 15;
  int ktile = i >> 12;
  int n = ntile * 16 + (lane & 15);
  int k = ktile * 32 + (pair >> 2) * 16 + (lane >> 4) * 8 + (pair & 3) * 2;
  Wb[i] = pack2_bf16(W_a[n * K_DIM + k], W_a[n * K_DIM + k + 1]);
}

// --------------------------- kernel 3: edge GEMM + score + scatter ----------
__launch_bounds__(128)
__global__ void k_gemm(const int*   __restrict__ trip,     // [E,3] (src,dst,rel)
                       const float* __restrict__ ent,      // [N_ENT,256]
                       const float* __restrict__ rel,      // [N_REL,256]
                       const float* __restrict__ b_a,      // [256]
                       const float* __restrict__ W2,       // [256]
                       const float* __restrict__ b2,       // [1]
                       const unsigned int* __restrict__ Wb,
                       float* __restrict__ out,            // [N_ENT,256] accumulators
                       float* __restrict__ e_total)
{
  __shared__ unsigned int ldsB[2 * TILE_DWORDS];           // 32 KB, double buffer

  const int tid  = threadIdx.x;
  const int wave = tid >> 5;
  const int lane = tid & 31;
  const long edge0 = (long)blockIdx.x * 64 + wave * 16;
  const bool valid = (edge0 < N_EDGES);                    // E is a multiple of 16

  // Per-lane gathered row pointers (row m = lane&15; lanes 16-31 mirror 0-15).
  const int m = lane & 15;
  const float* rowp0 = ent;
  const float* rowp1 = rel;
  const float* rowp2 = ent;
  int src_idx = 0;
  if (valid) {
    long e = edge0 + m;
    int t0 = trip[e * 3 + 0];
    int t1 = trip[e * 3 + 1];
    int t2 = trip[e * 3 + 2];
    rowp0 = ent + (long)t0 * IN_DIM;
    rowp1 = rel + (long)t2 * IN_DIM;
    rowp2 = ent + (long)t1 * IN_DIM;
    src_idx = t0;
  }

  v8f acc[N_TILES];
#pragma unroll
  for (int t = 0; t < N_TILES; ++t)
#pragma unroll
    for (int j = 0; j < 8; ++j) acc[t][j] = 0.f;

  const int khalf_off = (lane >> 4) * 8;   // A-fragment K sub-offset per lane half
  const unsigned ldsBase = (unsigned)(size_t)(void*)ldsB;

  // Prologue: TDM DMA of k-tile 0 into buffer 0 (issued by wave 0 only).
  if (wave == 0) tdm_load_16k(Wb, ldsBase);

  // A-gather prefetch for k-tile 0 (overlaps the TDM + first barrier).
  float4 ga0, ga1, ga2, ga3;
  if (valid) {
    ga0 = *(const float4*)(rowp0 + khalf_off);
    ga1 = *(const float4*)(rowp0 + khalf_off + 4);
    ga2 = *(const float4*)(rowp0 + khalf_off + 16);
    ga3 = *(const float4*)(rowp0 + khalf_off + 20);
  }

#pragma unroll 1
  for (int kt = 0; kt < K_TILES; ++kt) {
    if (wave == 0) {
      if (kt + 1 < K_TILES) {
        // Overlap: DMA next k-tile while this one is consumed.
        tdm_load_16k(Wb + (kt + 1) * TILE_DWORDS,
                     ldsBase + (unsigned)(((kt + 1) & 1) * (TILE_DWORDS * 4)));
        __builtin_amdgcn_s_wait_tensorcnt(1);   // tile kt has landed
      } else {
        __builtin_amdgcn_s_wait_tensorcnt(0);
      }
    }
    __syncthreads();                            // publish buffer kt&1

    if (valid) {
      // Pack this k-tile's A fragment from prefetched registers.
      union { v8u u; v16bf b; } A;
      A.u[0] = pack2_bf16(ga0.x, ga0.y); A.u[1] = pack2_bf16(ga0.z, ga0.w);
      A.u[2] = pack2_bf16(ga1.x, ga1.y); A.u[3] = pack2_bf16(ga1.z, ga1.w);
      A.u[4] = pack2_bf16(ga2.x, ga2.y); A.u[5] = pack2_bf16(ga2.z, ga2.w);
      A.u[6] = pack2_bf16(ga3.x, ga3.y); A.u[7] = pack2_bf16(ga3.z, ga3.w);

      // Prefetch A for k-tile kt+1: latency hides behind the 16 WMMAs below.
      if (kt + 1 < K_TILES) {
        const int kn = kt + 1;
        const float* rp = (kn < 8) ? rowp0 : ((kn < 16) ? rowp1 : rowp2);
        const int kb = (kn & 7) * 32 + khalf_off;
        ga0 = *(const float4*)(rp + kb);
        ga1 = *(const float4*)(rp + kb + 4);
        ga2 = *(const float4*)(rp + kb + 16);
        ga3 = *(const float4*)(rp + kb + 20);
      }

      const unsigned int* Bbase = ldsB + (kt & 1) * TILE_DWORDS;
      // Explicit ping-pong B fragments: load nt+1 before WMMA nt.
      union { v8u u; v16bf b; } B0, B1;
      load_bfrag(Bbase + lane * 8, B0.u);
#pragma unroll
      for (int nt = 0; nt < N_TILES; nt += 2) {
        load_bfrag(Bbase + ((nt + 1) * 32 + lane) * 8, B1.u);
        acc[nt] = __builtin_amdgcn_wmma_f32_16x16x32_bf16(
            false, A.b, false, B0.b, (short)0, acc[nt], false, false);
        if (nt + 2 < N_TILES)
          load_bfrag(Bbase + ((nt + 2) * 32 + lane) * 8, B0.u);
        acc[nt + 1] = __builtin_amdgcn_wmma_f32_16x16x32_bf16(
            false, A.b, false, B1.b, (short)0, acc[nt + 1], false, false);
      }
    }
    __syncthreads();                            // all reads of buffer kt&1 done
  }

  if (!valid) return;

  // ---- epilogue: c += b_a; b = leaky_relu(c.W2 + b2); e = exp(b) ----------
  // C layout: acc[t][r] holds c[row, n] with row = (lane<16 ? r : r+8),
  //           n = t*16 + (lane&15).
  const int nl = lane & 15;
  float rowpart[8];
#pragma unroll
  for (int r = 0; r < 8; ++r) rowpart[r] = 0.f;

#pragma unroll
  for (int t = 0; t < N_TILES; ++t) {
    float bav = b_a[t * 16 + nl];
    float w2v = W2[t * 16 + nl];
#pragma unroll
    for (int r = 0; r < 8; ++r) {
      float c = acc[t][r] + bav;
      acc[t][r] = c;
      rowpart[r] += c * w2v;
    }
  }

  const float b2v = b2[0];
  float ev[8];
  float wave_sum = 0.f;
#pragma unroll
  for (int r = 0; r < 8; ++r) {
    float p = rowpart[r];
    p += __shfl_xor(p, 1, 32);         // reduce over the 16-lane half (N dim)
    p += __shfl_xor(p, 2, 32);
    p += __shfl_xor(p, 4, 32);
    p += __shfl_xor(p, 8, 32);
    float bb = p + b2v;
    bb = (bb > 0.f) ? bb : 0.01f * bb; // leaky_relu(0.01)
    float e = __expf(bb);
    ev[r] = e;                          // per-half replicated, matches C rows
    wave_sum += __shfl(e, 0, 32) + __shfl(e, 16, 32);
  }
  if (lane == 0) atomicAdd(e_total, wave_sum);

  // ---- scatter: out[src[row], n] += e[row] * c[row, n] (segment_sum) ------
  int sv[8];
#pragma unroll
  for (int r = 0; r < 8; ++r)
    sv[r] = __shfl(src_idx, (lane < 16) ? r : (r + 8), 32);

#pragma unroll
  for (int t = 0; t < N_TILES; ++t) {
    const int n = t * 16 + nl;
#pragma unroll
    for (int r = 0; r < 8; ++r) {
      atomicAdd(out + (long)sv[r] * OUT_DIM + n, acc[t][r] * ev[r]);
    }
  }
}

// --------------------------- kernel 4: out = elu(out / e_total) -------------
__global__ void k_final(float4* __restrict__ out4, const float* __restrict__ e_total, int n4) {
  int i = blockIdx.x * blockDim.x + threadIdx.x;
  if (i >= n4) return;
  const float inv = 1.0f / *e_total;
  float4 v = out4[i];
  float x;
  x = v.x * inv; v.x = (x > 0.f) ? x : (__expf(x) - 1.f);
  x = v.y * inv; v.y = (x > 0.f) ? x : (__expf(x) - 1.f);
  x = v.z * inv; v.z = (x > 0.f) ? x : (__expf(x) - 1.f);
  x = v.w * inv; v.w = (x > 0.f) ? x : (__expf(x) - 1.f);
  out4[i] = v;
}

// ---------------------------------------------------------------------------
extern "C" void kernel_launch(void* const* d_in, const int* in_sizes, int n_in,
                              void* d_out, int out_size, void* d_ws, size_t ws_size,
                              hipStream_t stream) {
  const int*   trip  = (const int*)d_in[0];    // triplets [E,3] int32
  const float* ent   = (const float*)d_in[1];  // [N_ENT, 256]
  const float* rel   = (const float*)d_in[2];  // [N_REL, 256]
  const float* W_a   = (const float*)d_in[3];  // [256, 768]
  const float* b_a   = (const float*)d_in[4];  // [256]
  const float* W_a2  = (const float*)d_in[5];  // [1, 256]
  const float* b_a2  = (const float*)d_in[6];  // [1]
  float* out = (float*)d_out;                  // [N_ENT, 256]

  // Workspace: [0..4): e_b_total scalar; [256 ..): packed bf16 W (393 KB).
  float* e_total = (float*)d_ws;
  unsigned int* Wb = (unsigned int*)((char*)d_ws + 256);

  const int n4 = N_ENT * OUT_DIM / 4;          // 6.4M float4
  k_zero<<<(n4 + 255) / 256, 256, 0, stream>>>((float4*)out, e_total, n4);
  k_pack<<<(WB_DWORDS + 255) / 256, 256, 0, stream>>>(W_a, Wb);
  k_gemm<<<(N_EDGES + 63) / 64, 128, 0, stream>>>(trip, ent, rel, b_a, W_a2, b_a2,
                                                  Wb, out, e_total);
  k_final<<<(n4 + 255) / 256, 256, 0, stream>>>((float4*)out, e_total, n4);
}
